// GQA_90984587198870
// MI455X (gfx1250) — compile-verified, matrix operand order checked by reading, not run on hardware
//
#include <hip/hip_runtime.h>
#include <hip/hip_bf16.h>
#include <cstdint>

// ---------------------------------------------------------------------------
// GQA / MHA forward for gfx1250 (MI455X): bf16 WMMA + async LDS staging.
//   B=4, S=1024, HID=2048, H=16 heads, head_dim=128 (SUBH=1 -> plain MHA)
// Pipeline:
//   1) f32 -> bf16 conversion of x and weights
//   2) Q/K/V projections: bf16 WMMA GEMM (double-buffered async-LDS), f32 out
//   3) RoPE on Q,K (+ V convert) -> bf16
//   4) flash attention per (b, head), WMMA for QK^T and P*V
//   5) output projection: bf16 WMMA GEMM, f32 out (+bias) -> d_out
// ---------------------------------------------------------------------------

typedef __attribute__((ext_vector_type(16))) __bf16 v16bf;
typedef __attribute__((ext_vector_type(8)))  __bf16 v8bf;
typedef __attribute__((ext_vector_type(8)))  float  v8f;

#define WMMA_BF16(A, B, C) \
  __builtin_amdgcn_wmma_f32_16x16x32_bf16(false, (A), false, (B), (short)0, (C), false, false)

// --- gfx1250 async global->LDS copy (ASYNCcnt) with safe fallback ----------
#if defined(__gfx1250__) && __has_builtin(__builtin_amdgcn_global_load_async_to_lds_b128)
#define USE_ASYNC_LDS 1
#else
#define USE_ASYNC_LDS 0
#endif

#if USE_ASYNC_LDS
// Builtin signature (from hipcc diagnostic): param0 = AS(1) generic-vector
// int4* (global src), param1 = AS(3) int4* (LDS dst), then imm offset, imm cpol.
typedef int v4i_gcc __attribute__((vector_size(16)));
typedef __attribute__((address_space(1))) v4i_gcc* as1_v4i_p;
typedef __attribute__((address_space(3))) v4i_gcc* as3_v4i_p;
#endif

__device__ __forceinline__ void async_cp16(const __bf16* g, __bf16* l) {
#if USE_ASYNC_LDS
  __builtin_amdgcn_global_load_async_to_lds_b128((as1_v4i_p)g, (as3_v4i_p)l,
                                                 /*offset=*/0, /*cpol=*/0);
#else
  *(v8bf*)l = *(const v8bf*)g;
#endif
}

__device__ __forceinline__ void wait_async() {
#if USE_ASYNC_LDS
#if __has_builtin(__builtin_amdgcn_s_wait_asynccnt)
  __builtin_amdgcn_s_wait_asynccnt(0);
#else
  asm volatile("s_wait_asynccnt 0x0" ::: "memory");
#endif
#endif
}

// Load a 16x32 bf16 A-fragment (or column-major B-fragment) for
// v_wmma_f32_16x16x32_bf16.  Per ISA layout (16-bit A 16x32):
//   lane L: row = L%16, half h = L/16
//   elements 0..7  -> K = h*8 + e        (one contiguous 16B chunk)
//   elements 8..15 -> K = 16 + h*8 + e-8 (second contiguous 16B chunk)
__device__ __forceinline__ v16bf frag_ld(const __bf16* base, int stride, int lane) {
  const int rr = lane & 15;
  const int hh = lane >> 4;
  const __bf16* p = base + (size_t)rr * stride + hh * 8;
  v8bf lo = *(const v8bf*)(p);
  v8bf hi = *(const v8bf*)(p + 16);
  v16bf f;
#pragma unroll
  for (int i = 0; i < 8; ++i) { f[i] = lo[i]; f[i + 8] = hi[i]; }
  return f;
}

// ---------------------------------------------------------------------------
// 1) f32 -> bf16
// ---------------------------------------------------------------------------
__global__ __launch_bounds__(256) void f32_to_bf16_kernel(const float* __restrict__ in,
                                                          __bf16* __restrict__ out, int n) {
  int i = (blockIdx.x * 256 + threadIdx.x) * 4;
  if (i + 3 < n) {
    float4 v = *(const float4*)(in + i);
    out[i + 0] = (__bf16)v.x;
    out[i + 1] = (__bf16)v.y;
    out[i + 2] = (__bf16)v.z;
    out[i + 3] = (__bf16)v.w;
  }
}

// ---------------------------------------------------------------------------
// 2/5) C[m][n] = sum_k A[m][k] * W[n][k] + bias[n]      (A: MxK, W: NxK bf16)
// Block tile 128x128, BK=32, 8 waves; each wave owns a 32x64 region
// (2 M-tiles x 4 N-tiles of 16x16), 8 WMMAs per K-step.
// Double-buffered LDS tiles staged with global_load_async_to_lds_b128:
// copies for tile k+1 are in flight while WMMAs run on tile k.
// ---------------------------------------------------------------------------
__global__ __launch_bounds__(256) void gemm_bf16_kernel(const __bf16* __restrict__ A,
                                                        const __bf16* __restrict__ W,
                                                        const float* __restrict__ bias,
                                                        float* __restrict__ C,
                                                        int M, int N, int K) {
  __shared__ __bf16 As[2][128][40];   // +8 pad: 80B row stride, 16B aligned chunks
  __shared__ __bf16 Ws[2][128][40];
  const int tid = threadIdx.x, lane = tid & 31, wid = tid >> 5;
  const int wm = wid & 3;             // 4 waves along M
  const int wn = wid >> 2;            // 2 waves along N
  const int m0 = blockIdx.y * 128, n0 = blockIdx.x * 128;

  // Per-thread staging slice: one row, 16 consecutive bf16 (two 16B chunks).
  const int ldRow = tid >> 1;         // 0..127
  const int ldCol = (tid & 1) * 16;   // 0 or 16

  v8f vzero = {};
  v8f acc[2][4];
#pragma unroll
  for (int t = 0; t < 2; ++t)
#pragma unroll
    for (int u = 0; u < 4; ++u) acc[t][u] = vzero;

  const __bf16* gA = A + (size_t)(m0 + ldRow) * K + ldCol;
  const __bf16* gW = W + (size_t)(n0 + ldRow) * K + ldCol;

  // prologue: stage tile 0 into buffer 0
  async_cp16(gA,     &As[0][ldRow][ldCol]);
  async_cp16(gA + 8, &As[0][ldRow][ldCol + 8]);
  async_cp16(gW,     &Ws[0][ldRow][ldCol]);
  async_cp16(gW + 8, &Ws[0][ldRow][ldCol + 8]);
  wait_async();
  __syncthreads();

  int buf = 0;
  for (int k0 = 0; k0 < K; k0 += 32) {
    if (k0 + 32 < K) {                // issue async copies for the next tile
      const __bf16* nA = gA + k0 + 32;
      const __bf16* nW = gW + k0 + 32;
      async_cp16(nA,     &As[buf ^ 1][ldRow][ldCol]);
      async_cp16(nA + 8, &As[buf ^ 1][ldRow][ldCol + 8]);
      async_cp16(nW,     &Ws[buf ^ 1][ldRow][ldCol]);
      async_cp16(nW + 8, &Ws[buf ^ 1][ldRow][ldCol + 8]);
#if !USE_ASYNC_LDS
    }
    if (k0 + 64 < K) {                // fallback path: prefetch tile k+2
      __builtin_prefetch(gA + k0 + 64, 0, 1);
      __builtin_prefetch(gW + k0 + 64, 0, 1);
#endif
    }

    v16bf af[2], bfr[4];
#pragma unroll
    for (int t = 0; t < 2; ++t) af[t] = frag_ld(&As[buf][wm * 32 + t * 16][0], 40, lane);
#pragma unroll
    for (int u = 0; u < 4; ++u) bfr[u] = frag_ld(&Ws[buf][wn * 64 + u * 16][0], 40, lane);
#pragma unroll
    for (int t = 0; t < 2; ++t)
#pragma unroll
      for (int u = 0; u < 4; ++u)
        acc[t][u] = WMMA_BF16(af[t], bfr[u], acc[t][u]);

    wait_async();                     // this wave's copies into buf^1 landed
    __syncthreads();                  // everyone done reading buf / writing buf^1
    buf ^= 1;
  }

  // C/D layout: lane L holds col n=L%16, rows m=(L/16)*8+v in vgpr v.
  const int rr = lane & 15, hh = lane >> 4;
#pragma unroll
  for (int t = 0; t < 2; ++t)
#pragma unroll
    for (int u = 0; u < 4; ++u) {
      int col = n0 + wn * 64 + u * 16 + rr;
      float bv = bias[col];
#pragma unroll
      for (int v = 0; v < 8; ++v) {
        int row = m0 + wm * 32 + t * 16 + hh * 8 + v;
        C[(size_t)row * N + col] = acc[t][u][v] + bv;
      }
    }
}

// ---------------------------------------------------------------------------
// 3) RoPE on Q,K (f32 in, bf16 out) + V convert. One thread per rotation pair.
// ---------------------------------------------------------------------------
__global__ __launch_bounds__(256) void rope_cvt_kernel(const float* __restrict__ Qf,
                                                       const float* __restrict__ Kf,
                                                       const float* __restrict__ Vf,
                                                       __bf16* __restrict__ Qb,
                                                       __bf16* __restrict__ Kb,
                                                       __bf16* __restrict__ Vb) {
  int t = blockIdx.x * 256 + threadIdx.x;   // B*S*H*64 = 2^22 threads
  int i    = t & 63;
  int head = (t >> 6) & 15;
  int s    = (t >> 10) & 1023;
  int b    = t >> 20;
  size_t idx1 = ((size_t)(b * 1024 + s)) * 2048 + head * 128 + i;
  size_t idx2 = idx1 + 64;
  float ang = (float)s * __powf(10000.0f, -(float)i * (1.0f / 64.0f));
  float c = __cosf(ang), sn = __sinf(ang);
  float q1 = Qf[idx1], q2 = Qf[idx2];
  Qb[idx1] = (__bf16)(q1 * c - q2 * sn);
  Qb[idx2] = (__bf16)(q1 * sn + q2 * c);
  float k1 = Kf[idx1], k2 = Kf[idx2];
  Kb[idx1] = (__bf16)(k1 * c - k2 * sn);
  Kb[idx2] = (__bf16)(k1 * sn + k2 * c);
  Vb[idx1] = (__bf16)Vf[idx1];
  Vb[idx2] = (__bf16)Vf[idx2];
}

// ---------------------------------------------------------------------------
// 4) Flash attention, one block per (b, head, 128-row query tile).
//    8 waves x 16 query rows; key tiles of 64, online softmax, WMMA for
//    QK^T and P*V.  K tile staged with async global->LDS copies; V is
//    transposed through VGPRs into LDS so the P*V B-fragment loads
//    contiguous keys.  P goes through per-wave LDS scratch (D-layout ->
//    A-layout conversion); in-wave DS ops are in-order so no barrier needed.
// ---------------------------------------------------------------------------
__global__ __launch_bounds__(256) void attn_kernel(const __bf16* __restrict__ Q,
                                                   const __bf16* __restrict__ Kk,
                                                   const __bf16* __restrict__ V,
                                                   __bf16* __restrict__ O) {
  constexpr int S = 1024, HID = 2048, D = 128, BN = 64;
  __shared__ __bf16 Ks[BN][D + 8];        // keys row-major (key x dim)
  __shared__ __bf16 Vt[D][BN + 8];        // V transposed (dim x key)
  __shared__ __bf16 Ps[8][16][BN + 8];    // per-wave P scratch

  const int tid = threadIdx.x, lane = tid & 31, wid = tid >> 5;
  const int rr = lane & 15, hh = lane >> 4;
  const int b = blockIdx.x >> 4, h = blockIdx.x & 15;
  const size_t headBase = ((size_t)b * S) * HID + (size_t)h * D;
  const int qrow0 = blockIdx.y * 128 + wid * 16;

  // Q fragments held in registers for the whole kernel: 16x128 per wave.
  v16bf qf[4];
#pragma unroll
  for (int f = 0; f < 4; ++f)
    qf[f] = frag_ld(Q + headBase + (size_t)qrow0 * HID + f * 32, HID, lane);

  v8f vzero = {};
  v8f o_acc[8];
#pragma unroll
  for (int ot = 0; ot < 8; ++ot) o_acc[ot] = vzero;
  float m_run[8], l_run[8];
#pragma unroll
  for (int v = 0; v < 8; ++v) { m_run[v] = -1e30f; l_run[v] = 0.0f; }
  const float scale = 0.088388347648318447f;  // 1/sqrt(128)

  for (int j = 0; j < S / BN; ++j) {
    // Stage K tile (async -> LDS) and transposed V tile (through VGPRs).
    for (int c = tid; c < BN * (D / 8); c += 256) {
      int kr = c >> 4, cc = (c & 15) * 8;
      async_cp16(Kk + headBase + (size_t)(j * BN + kr) * HID + cc, &Ks[kr][cc]);
      v8bf vv = *(const v8bf*)(V + headBase + (size_t)(j * BN + kr) * HID + cc);
#pragma unroll
      for (int e = 0; e < 8; ++e) Vt[cc + e][kr] = vv[e];
    }
    wait_async();
    __syncthreads();

    // scores = Q * K^T * scale : 4 n-tiles of 16 keys, K-dim = 128.
    v8f sf[4];
#pragma unroll
    for (int nt = 0; nt < 4; ++nt) {
      v8f a = vzero;
#pragma unroll
      for (int f = 0; f < 4; ++f) {
        v16bf kb = frag_ld(&Ks[nt * 16][f * 32], D + 8, lane);
        a = WMMA_BF16(qf[f], kb, a);
      }
      sf[nt] = a * scale;
    }

    // row max across 64 keys (row m = hh*8+v lives in 16 lanes of one half).
    float mx[8];
#pragma unroll
    for (int v = 0; v < 8; ++v) {
      float m = sf[0][v];
#pragma unroll
      for (int nt = 1; nt < 4; ++nt) m = fmaxf(m, sf[nt][v]);
      for (int off = 1; off < 16; off <<= 1) m = fmaxf(m, __shfl_xor(m, off, 32));
      mx[v] = m;
    }

    // online softmax update
    float corr[8], rs[8];
#pragma unroll
    for (int v = 0; v < 8; ++v) {
      float mn = fmaxf(m_run[v], mx[v]);
      corr[v] = __expf(m_run[v] - mn);
      m_run[v] = mn;
      rs[v] = 0.0f;
    }
#pragma unroll
    for (int nt = 0; nt < 4; ++nt)
#pragma unroll
      for (int v = 0; v < 8; ++v) {
        float p = __expf(sf[nt][v] - m_run[v]);
        sf[nt][v] = p;
        rs[v] += p;
      }
#pragma unroll
    for (int v = 0; v < 8; ++v) {
      for (int off = 1; off < 16; off <<= 1) rs[v] += __shfl_xor(rs[v], off, 32);
      l_run[v] = l_run[v] * corr[v] + rs[v];
    }
#pragma unroll
    for (int ot = 0; ot < 8; ++ot)
#pragma unroll
      for (int v = 0; v < 8; ++v) o_acc[ot][v] *= corr[v];

    // P (D-layout) -> per-wave LDS -> A-layout fragments
#pragma unroll
    for (int nt = 0; nt < 4; ++nt)
#pragma unroll
      for (int v = 0; v < 8; ++v)
        Ps[wid][hh * 8 + v][nt * 16 + rr] = (__bf16)sf[nt][v];

    // O += P * V : A = P (16x64), B = V (64 keys x 16 dims) from Vt.
#pragma unroll
    for (int f = 0; f < 2; ++f) {
      v16bf pf = frag_ld(&Ps[wid][0][f * 32], BN + 8, lane);
#pragma unroll
      for (int ot = 0; ot < 8; ++ot) {
        v16bf vf = frag_ld(&Vt[ot * 16][f * 32], BN + 8, lane);
        o_acc[ot] = WMMA_BF16(pf, vf, o_acc[ot]);
      }
    }
    __syncthreads();
  }

  // normalize and store bf16 attention output
#pragma unroll
  for (int ot = 0; ot < 8; ++ot)
#pragma unroll
    for (int v = 0; v < 8; ++v) {
      int row = qrow0 + hh * 8 + v;
      int col = ot * 16 + rr;
      O[headBase + (size_t)row * HID + col] = (__bf16)(o_acc[ot][v] / l_run[v]);
    }
}

// ---------------------------------------------------------------------------
// host driver
// ---------------------------------------------------------------------------
extern "C" void kernel_launch(void* const* d_in, const int* in_sizes, int n_in,
                              void* d_out, int out_size, void* d_ws, size_t ws_size,
                              hipStream_t stream) {
  (void)in_sizes; (void)n_in; (void)out_size; (void)ws_size;
  const float* x  = (const float*)d_in[0];
  const float* Wq = (const float*)d_in[1];
  const float* bq = (const float*)d_in[2];
  const float* Wk = (const float*)d_in[3];
  const float* bk = (const float*)d_in[4];
  const float* Wv = (const float*)d_in[5];
  const float* bv = (const float*)d_in[6];
  const float* Wo = (const float*)d_in[7];
  const float* bo = (const float*)d_in[8];

  constexpr int Bb = 4, S = 1024, HID = 2048;
  constexpr size_t Mrows = (size_t)Bb * S;     // 4096
  constexpr size_t nX = Mrows * HID;           // 8,388,608
  constexpr size_t nW = (size_t)HID * HID;     // 4,194,304

  char* ws = (char*)d_ws;
  size_t off = 0;
  auto take = [&](size_t bytes) -> char* {
    char* p = ws + off;
    off = (off + bytes + 255) & ~(size_t)255;
    return p;
  };
  __bf16* xb  = (__bf16*)take(nX * 2);
  __bf16* Wqb = (__bf16*)take(nW * 2);
  __bf16* Wkb = (__bf16*)take(nW * 2);
  __bf16* Wvb = (__bf16*)take(nW * 2);
  __bf16* Wob = (__bf16*)take(nW * 2);
  float*  Qf  = (float*)take(nX * 4);
  float*  Kf  = (float*)take(nX * 4);
  float*  Vf  = (float*)take(nX * 4);
  __bf16* Qb  = (__bf16*)take(nX * 2);
  __bf16* Kb  = (__bf16*)take(nX * 2);
  __bf16* Vb  = (__bf16*)take(nX * 2);
  __bf16* AOb = (__bf16*)Qf;   // overlay: Qf is dead after rope_cvt_kernel

  // 1) convert inputs/weights to bf16
  f32_to_bf16_kernel<<<(int)(nX / 4 / 256), 256, 0, stream>>>(x,  xb,  (int)nX);
  f32_to_bf16_kernel<<<(int)(nW / 4 / 256), 256, 0, stream>>>(Wq, Wqb, (int)nW);
  f32_to_bf16_kernel<<<(int)(nW / 4 / 256), 256, 0, stream>>>(Wk, Wkb, (int)nW);
  f32_to_bf16_kernel<<<(int)(nW / 4 / 256), 256, 0, stream>>>(Wv, Wvb, (int)nW);
  f32_to_bf16_kernel<<<(int)(nW / 4 / 256), 256, 0, stream>>>(Wo, Wob, (int)nW);

  // 2) Q/K/V projections (WMMA GEMM, f32 out)
  dim3 gGemm(HID / 128, (unsigned)(Mrows / 128));  // (16, 32)
  gemm_bf16_kernel<<<gGemm, 256, 0, stream>>>(xb, Wqb, bq, Qf, (int)Mrows, HID, HID);
  gemm_bf16_kernel<<<gGemm, 256, 0, stream>>>(xb, Wkb, bk, Kf, (int)Mrows, HID, HID);
  gemm_bf16_kernel<<<gGemm, 256, 0, stream>>>(xb, Wvb, bv, Vf, (int)Mrows, HID, HID);

  // 3) RoPE + bf16 conversion
  rope_cvt_kernel<<<(Bb * S * 16 * 64) / 256, 256, 0, stream>>>(Qf, Kf, Vf, Qb, Kb, Vb);

  // 4) attention (WMMA flash attention)
  attn_kernel<<<dim3(Bb * 16, S / 128), 256, 0, stream>>>(Qb, Kb, Vb, AOb);

  // 5) output projection -> f32 d_out
  gemm_bf16_kernel<<<gGemm, 256, 0, stream>>>(AOb, Wob, bo, (float*)d_out, (int)Mrows, HID, HID);
}
